// BrainInspiredRouter_8315056685630
// MI455X (gfx1250) — compile-verified, matrix operand order checked by reading, not run on hardware
//
#include <hip/hip_runtime.h>

// Problem constants (match reference)
#define B_    16384
#define D_    1024
#define DOUT_ 1024
#define R_    8
#define DH_   512
#define NH_   (R_ * DH_)   // 4096

typedef __attribute__((ext_vector_type(16))) __bf16 v16bf;
typedef __attribute__((ext_vector_type(8)))  __bf16 v8bf;
typedef __attribute__((ext_vector_type(8)))  float  v8f;
typedef int v4i_g __attribute__((vector_size(16)));   // matches builtin's V4i param

#define GLOBAL_AS __attribute__((address_space(1)))
#define LDS_AS    __attribute__((address_space(3)))

#if defined(__AMDGCN__) && __has_builtin(__builtin_amdgcn_global_load_async_to_lds_b128)
#define HAVE_ASYNC_LDS 1
#else
#define HAVE_ASYNC_LDS 0
#endif

#if HAVE_ASYNC_LDS
#define ASYNC_CP128(gsrc, ldst) \
  __builtin_amdgcn_global_load_async_to_lds_b128((GLOBAL_AS v4i_g*)(gsrc), (LDS_AS v4i_g*)(ldst), 0, 0)
#endif

__device__ __forceinline__ void wait_async0() {
#if __has_builtin(__builtin_amdgcn_s_wait_asynccnt)
  __builtin_amdgcn_s_wait_asynccnt(0);
#else
  asm volatile("s_wait_asynccnt 0x0" ::: "memory");
#endif
}

__device__ __forceinline__ unsigned short f2bf(float f) {
  unsigned int u = __float_as_uint(f);
  u += 0x7fffu + ((u >> 16) & 1u);        // round-to-nearest-even
  return (unsigned short)(u >> 16);
}

// ---------------- f32 -> bf16 convert (vector x4) ----------------
__global__ void k_cvt_bf16(const float* __restrict__ s, unsigned short* __restrict__ d, long n) {
  long i = ((long)blockIdx.x * blockDim.x + threadIdx.x) * 4;
  if (i >= n) return;
  float4 v = *(const float4*)(s + i);
  ushort4 o;
  o.x = f2bf(v.x); o.y = f2bf(v.y); o.z = f2bf(v.z); o.w = f2bf(v.w);
  *(ushort4*)(d + i) = o;
}

// ---------------- W2 (R,DOUT,DH) -> W2cat (DOUT, R*DH) bf16 ----------------
__global__ void k_perm_w2(const float* __restrict__ s, unsigned short* __restrict__ d) {
  long idx = (long)blockIdx.x * blockDim.x + threadIdx.x;   // < R*DOUT*DH
  int k  = (int)(idx & (DH_ - 1));
  long t2 = idx >> 9;
  int o  = (int)(t2 & (DOUT_ - 1));
  int r  = (int)(t2 >> 10);
  d[(long)o * NH_ + (long)r * DH_ + k] = f2bf(s[idx]);
}

// ---------------- gate = softmax(x @ Wg.T + bg), one wave per row ----------------
__global__ __launch_bounds__(128) void k_gate(const float* __restrict__ x,
                                              const float* __restrict__ Wg,
                                              const float* __restrict__ bg,
                                              float* __restrict__ gate) {
  __shared__ float sWg[R_ * D_];
  for (int i = threadIdx.x; i < R_ * D_; i += 128) sWg[i] = Wg[i];
  __syncthreads();
  const int lane = threadIdx.x & 31;
  const int b = blockIdx.x * 4 + (threadIdx.x >> 5);
  const float* xr = x + (size_t)b * D_;
  float acc[R_];
#pragma unroll
  for (int q = 0; q < R_; ++q) acc[q] = 0.f;
  for (int d = lane; d < D_; d += 32) {
    float xv = xr[d];
#pragma unroll
    for (int q = 0; q < R_; ++q) acc[q] += xv * sWg[q * D_ + d];
  }
#pragma unroll
  for (int o2 = 16; o2 > 0; o2 >>= 1)
#pragma unroll
    for (int q = 0; q < R_; ++q) acc[q] += __shfl_xor(acc[q], o2, 32);
  if (lane == 0) {
    float mx = -3.0e38f;
#pragma unroll
    for (int q = 0; q < R_; ++q) { acc[q] += bg[q]; mx = fmaxf(mx, acc[q]); }
    float sum = 0.f, e[R_];
#pragma unroll
    for (int q = 0; q < R_; ++q) { e[q] = __expf(acc[q] - mx); sum += e[q]; }
    float inv = 1.f / sum;
#pragma unroll
    for (int q = 0; q < R_; ++q) gate[(size_t)b * R_ + q] = e[q] * inv;
  }
}

// ---------------- bf16 WMMA GEMM: C(MxN) = A(MxK) @ W(NxK)^T ----------------
// MODE 0/1: +bias[n],                       store bf16
// MODE 2:   +bias[n], relu, *gate[m,n>>9],  store bf16
// MODE 3:   + sum_r gate[m,r]*bias[r*N+n],  store f32
#define BM   128
#define BN   128
#define BK   32
#define LDST 40   // padded LDS row stride (bf16 elems): 80B, conflict-free b128 reads

template <int MODE>
__global__ __launch_bounds__(256) void k_gemm(const unsigned short* __restrict__ A,
                                              const unsigned short* __restrict__ W,
                                              const float* __restrict__ bias,
                                              const float* __restrict__ gate,
                                              void* __restrict__ Out,
                                              int M, int N, int K) {
  __shared__ __align__(16) unsigned short sA[2][BM * LDST];
  __shared__ __align__(16) unsigned short sB[2][BN * LDST];

  const int t  = threadIdx.x;
  const int m0 = blockIdx.y * BM;
  const int n0 = blockIdx.x * BN;

  // global -> LDS staging: 2 x b128 per operand per thread per K-step
  const int lr = t >> 2;            // 0..63
  const int lc = (t & 3) * 8;       // 0,8,16,24 (bf16 elems)
  const unsigned short* ga0 = A + (size_t)(m0 + lr) * K + lc;
  const unsigned short* ga1 = A + (size_t)(m0 + 64 + lr) * K + lc;
  const unsigned short* gb0 = W + (size_t)(n0 + lr) * K + lc;
  const unsigned short* gb1 = W + (size_t)(n0 + 64 + lr) * K + lc;
  const int so0 = lr * LDST + lc;
  const int so1 = (lr + 64) * LDST + lc;

  const int lane = t & 31;
  const int rr   = lane & 15;       // row within 16x16 subtile
  const int kg   = lane >> 4;       // K-group (0/1)
  const int wm   = ((t >> 5) >> 2) * 64;   // wave M offset: 0/64
  const int wn   = ((t >> 5) & 3) * 32;    // wave N offset: 0..96

  v8f acc[4][2] = {};
  const int KT = K / BK;

#if HAVE_ASYNC_LDS
  // ---- async global->LDS fill path (ASYNCcnt) ----
  ASYNC_CP128(ga0, &sA[0][so0]);
  ASYNC_CP128(ga1, &sA[0][so1]);
  ASYNC_CP128(gb0, &sB[0][so0]);
  ASYNC_CP128(gb1, &sB[0][so1]);
  wait_async0();
  __syncthreads();
#else
  uint4 ra0 = *(const uint4*)ga0;
  uint4 ra1 = *(const uint4*)ga1;
  uint4 rb0 = *(const uint4*)gb0;
  uint4 rb1 = *(const uint4*)gb1;
  *(uint4*)&sA[0][so0] = ra0;
  *(uint4*)&sA[0][so1] = ra1;
  *(uint4*)&sB[0][so0] = rb0;
  *(uint4*)&sB[0][so1] = rb1;
  __syncthreads();
#endif

  for (int kt = 0; kt < KT; ++kt) {
    const int cur = kt & 1;
#if HAVE_ASYNC_LDS
    if (kt + 1 < KT) {
      const int nb = cur ^ 1;
      const size_t ko = (size_t)(kt + 1) * BK;
      ASYNC_CP128(ga0 + ko, &sA[nb][so0]);
      ASYNC_CP128(ga1 + ko, &sA[nb][so1]);
      ASYNC_CP128(gb0 + ko, &sB[nb][so0]);
      ASYNC_CP128(gb1 + ko, &sB[nb][so1]);
    }
#else
    if (kt + 1 < KT) {
      const size_t ko = (size_t)(kt + 1) * BK;
      ra0 = *(const uint4*)(ga0 + ko);
      ra1 = *(const uint4*)(ga1 + ko);
      rb0 = *(const uint4*)(gb0 + ko);
      rb1 = *(const uint4*)(gb1 + ko);
    }
#endif
    const unsigned short* bA = sA[cur];
    const unsigned short* bB = sB[cur];
    v16bf af[4], bfr[2];
#pragma unroll
    for (int i = 0; i < 4; ++i) {
      const unsigned short* p = bA + (wm + i * 16 + rr) * LDST + kg * 8;
      v8bf lo = *(const v8bf*)p;
      v8bf hi = *(const v8bf*)(p + 16);
      af[i] = __builtin_shufflevector(lo, hi, 0,1,2,3,4,5,6,7,8,9,10,11,12,13,14,15);
    }
#pragma unroll
    for (int j = 0; j < 2; ++j) {
      const unsigned short* p = bB + (wn + j * 16 + rr) * LDST + kg * 8;
      v8bf lo = *(const v8bf*)p;
      v8bf hi = *(const v8bf*)(p + 16);
      bfr[j] = __builtin_shufflevector(lo, hi, 0,1,2,3,4,5,6,7,8,9,10,11,12,13,14,15);
    }
#pragma unroll
    for (int i = 0; i < 4; ++i)
#pragma unroll
      for (int j = 0; j < 2; ++j)
        acc[i][j] = __builtin_amdgcn_wmma_f32_16x16x32_bf16(
            false, af[i], false, bfr[j], (short)0, acc[i][j], false, false);
#if HAVE_ASYNC_LDS
    if (kt + 1 < KT) wait_async0();
    __syncthreads();
#else
    if (kt + 1 < KT) {
      const int nb = cur ^ 1;
      *(uint4*)&sA[nb][so0] = ra0;
      *(uint4*)&sA[nb][so1] = ra1;
      *(uint4*)&sB[nb][so0] = rb0;
      *(uint4*)&sB[nb][so1] = rb1;
    }
    __syncthreads();
#endif
  }

  // ---- epilogue (C layout: VGPR e -> row kg*8+e, lane&15 -> col) ----
  float* outF = (float*)Out;
  unsigned short* outB = (unsigned short*)Out;
#pragma unroll
  for (int j = 0; j < 2; ++j) {
    const int col = n0 + wn + j * 16 + rr;
    float bcol = 0.f;
    float b2v[8];
    if (MODE <= 2) {
      bcol = bias[col];
    } else {
#pragma unroll
      for (int q = 0; q < 8; ++q) b2v[q] = bias[q * N + col];
    }
#pragma unroll
    for (int i = 0; i < 4; ++i) {
      const int rbase = m0 + wm + i * 16 + kg * 8;
#pragma unroll
      for (int e = 0; e < 8; ++e) {
        const int row = rbase + e;
        float cv = acc[i][j][e];
        if (MODE == 0 || MODE == 1) {
          outB[(size_t)row * N + col] = f2bf(cv + bcol);
        } else if (MODE == 2) {
          cv += bcol;
          cv = cv > 0.f ? cv : 0.f;
          cv *= gate[(size_t)row * 8 + (col >> 9)];
          outB[(size_t)row * N + col] = f2bf(cv);
        } else {
          float bb = 0.f;
#pragma unroll
          for (int q = 0; q < 8; ++q) bb += gate[(size_t)row * 8 + q] * b2v[q];
          outF[(size_t)row * N + col] = cv + bb;
        }
      }
    }
  }
}

extern "C" void kernel_launch(void* const* d_in, const int* in_sizes, int n_in,
                              void* d_out, int out_size, void* d_ws, size_t ws_size,
                              hipStream_t stream) {
  const float* x   = (const float*)d_in[0];
  const float* ipw = (const float*)d_in[1];
  const float* ipb = (const float*)d_in[2];
  const float* opw = (const float*)d_in[3];
  const float* opb = (const float*)d_in[4];
  const float* W1  = (const float*)d_in[5];
  const float* b1  = (const float*)d_in[6];
  const float* W2  = (const float*)d_in[7];
  const float* b2  = (const float*)d_in[8];
  const float* Wg  = (const float*)d_in[9];
  const float* bg  = (const float*)d_in[10];
  float* out = (float*)d_out;

  char* ws = (char*)d_ws;
  size_t off = 0;
  auto walloc = [&](size_t bytes) -> void* {
    void* p = ws + off;
    off = (off + bytes + 255) & ~(size_t)255;
    return p;
  };
  unsigned short* xb  = (unsigned short*)walloc((size_t)B_ * D_ * 2);
  unsigned short* vb  = (unsigned short*)walloc((size_t)B_ * D_ * 2);
  unsigned short* hg  = (unsigned short*)walloc((size_t)B_ * NH_ * 2);
  unsigned short* wvb = (unsigned short*)walloc((size_t)D_ * D_ * 2);
  unsigned short* wob = (unsigned short*)walloc((size_t)D_ * D_ * 2);
  unsigned short* w1b = (unsigned short*)walloc((size_t)NH_ * D_ * 2);
  unsigned short* w2b = (unsigned short*)walloc((size_t)DOUT_ * NH_ * 2);
  float* gate         = (float*)walloc((size_t)B_ * R_ * 4);
  unsigned short* ab  = xb;   // reuse x_bf16 buffer for 'attended' after GEMM0

  // conversions
  const long nx = (long)B_ * D_;
  k_cvt_bf16<<<(int)(nx / 4 / 256), 256, 0, stream>>>(x, xb, nx);
  const long nw = (long)D_ * D_;
  k_cvt_bf16<<<(int)(nw / 4 / 256), 256, 0, stream>>>(ipw + (size_t)2 * D_ * D_, wvb, nw);
  k_cvt_bf16<<<(int)(nw / 4 / 256), 256, 0, stream>>>(opw, wob, nw);
  const long n1 = (long)NH_ * D_;
  k_cvt_bf16<<<(int)(n1 / 4 / 256), 256, 0, stream>>>(W1, w1b, n1);
  k_perm_w2<<<(R_ * DOUT_ * DH_) / 256, 256, 0, stream>>>(W2, w2b);

  // gate (softmax over routes)
  k_gate<<<B_ / 4, 128, 0, stream>>>(x, Wg, bg, gate);

  // GEMM chain
  k_gemm<0><<<dim3(D_ / BN, B_ / BM), 256, 0, stream>>>(xb, wvb, ipb + 2 * D_, nullptr, vb, B_, D_, D_);
  k_gemm<1><<<dim3(D_ / BN, B_ / BM), 256, 0, stream>>>(vb, wob, opb, nullptr, ab, B_, D_, D_);
  k_gemm<2><<<dim3(NH_ / BN, B_ / BM), 256, 0, stream>>>(ab, w1b, b1, gate, hg, B_, NH_, D_);
  k_gemm<3><<<dim3(DOUT_ / BN, B_ / BM), 256, 0, stream>>>(hg, w2b, b2, gate, out, B_, DOUT_, NH_);
}
